// CFConv_37220186587164
// MI455X (gfx1250) — matrix operand order repaired and checked by reading, not compiled, not run on hardware
//
#include <hip/hip_runtime.h>
#include <hip/hip_bf16.h>

typedef _Float16 v16h __attribute__((ext_vector_type(16)));
typedef _Float16 v8h  __attribute__((ext_vector_type(8)));
typedef float    v8f  __attribute__((ext_vector_type(8)));

#define BGRAPH 8
#define NATOMS 256
#define FEAT   128
#define NTILES ((BGRAPH * NATOMS * NATOMS) / 16)   // 32768 row-tiles of 16 pairs

// shifted softplus: ln(1+e^x) - ln(2), numerically stable, raw-HW-transcendental form.
// 1 + exp(-|x|) is in (1,2]: never denormal / inf / NaN, so the raw v_log_f32
// (log2) needs no libm fixup code. ln(1+u) - ln2 == ln2 * (log2(1+u) - 1).
__device__ __forceinline__ float ssp(float x) {
    const float LOG2E = 1.4426950408889634f;
    const float LN2   = 0.6931471805599453f;
    float u = __builtin_amdgcn_exp2f(-LOG2E * fabsf(x));   // exp(-|x|) in (0,1]
    float l = __builtin_amdgcn_logf(1.0f + u);             // log2 of (1,2]
    return fmaxf(x, 0.0f) + LN2 * (l - 1.0f);
}

__global__ __launch_bounds__(256) void cfconv_fused_wmma(
    const float* __restrict__ pos,   // [B*N, 3]
    const float* __restrict__ w1,    // [1, F]
    const float* __restrict__ b1,    // [F]
    const float* __restrict__ w2,    // [F, F] row-major
    const float* __restrict__ b2,    // [F]
    const float* __restrict__ wd,    // [F, A] row-major
    const float* __restrict__ bd,    // [A]
    float* __restrict__ out)         // [M, A] f32
{
    // Weights pre-swizzled into WMMA B-fragment layout:
    //   fragment (kt, nt): lane L holds column nt*16 + (L&15),
    //   16 halves: K = kt*32 + (L<16 ? 0 : 8) + (e<8 ? e : e+8)
    __shared__ __attribute__((aligned(32))) _Float16 sW2[4 * 8 * 32 * 16];   // 32 KB
    __shared__ __attribute__((aligned(32))) _Float16 sWD[4 * 8 * 32 * 16];   // 32 KB
    __shared__ __attribute__((aligned(32))) _Float16 sStage[8][16 * FEAT];   // 32 KB (per-wave transpose tile)
    __shared__ float sW1[FEAT], sB1[FEAT], sB2[FEAT], sBD[FEAT];

    const int tid = threadIdx.x;

    // ---- one-time per-block: convert + swizzle weights into LDS ----
    for (int idx = tid; idx < 4 * 8 * 32; idx += 256) {
        const int fl = idx & 31;          // fragment lane
        const int nt = (idx >> 5) & 7;    // column tile
        const int kt = idx >> 8;          // k tile
        const int n  = nt * 16 + (fl & 15);
        const int kb = kt * 32 + ((fl & 16) ? 8 : 0);
        _Float16* d2 = &sW2[idx * 16];
        _Float16* dD = &sWD[idx * 16];
#pragma unroll
        for (int e = 0; e < 16; ++e) {
            d2[e] = (_Float16)w2[(kb + (e < 8 ? e : e + 8)) * FEAT + n];
            dD[e] = (_Float16)wd[(kb + (e < 8 ? e : e + 8)) * FEAT + n];
        }
    }
    if (tid < FEAT) {
        sW1[tid] = w1[tid];
        sB1[tid] = b1[tid];
        sB2[tid] = b2[tid];
        sBD[tid] = bd[tid];
    }
    __syncthreads();

    const int lane  = tid & 31;
    const int wave  = tid >> 5;
    const int r     = lane & 15;                 // row within 16-row tile
    const int hioff = (lane & 16) ? 8 : 0;       // K offset for upper half-wave (A layout)
    const int rowo  = (lane & 16) ? 8 : 0;       // row offset for upper half-wave (C/D layout)
    _Float16* stage = &sStage[wave][0];

    const int gw = blockIdx.x * 8 + wave;
    const int nw = gridDim.x * 8;

    for (int t = gw; t < NTILES; t += nw) {
        // ---- pairwise distance for this lane's row ----
        const int m  = t * 16 + r;
        const int b  = m >> 16;            // N*N = 65536
        const int ij = m & 0xFFFF;
        const int i  = ij >> 8;
        const int j  = ij & 0xFF;
        const float* pi = pos + (size_t)(b * NATOMS + i) * 3;
        const float* pj = pos + (size_t)(b * NATOMS + j) * 3;
        const float dx = pi[0] - pj[0];
        const float dy = pi[1] - pj[1];
        const float dz = pi[2] - pj[2];
        const float dist = sqrtf(dx * dx + dy * dy + dz * dz + 1e-12f);

        // ---- H in A-fragment layout: lane = row, halves = K ----
        // 16-bit A 16x32: e<8 -> K=e, e>=8 -> K=e+8; +8 for lanes 16..31
        v16h aH[4];
#pragma unroll
        for (int kt = 0; kt < 4; ++kt) {
#pragma unroll
            for (int e = 0; e < 16; ++e) {
                const int f = kt * 32 + (e < 8 ? e : e + 8) + hioff;
                aH[kt][e] = (_Float16)ssp(__builtin_fmaf(dist, sW1[f], sB1[f]));
            }
        }

        // ---- GEMM1: filters = ssp(H @ W2 + b2); stage as f16 row-major 16x128 ----
#pragma unroll
        for (int nt = 0; nt < 8; ++nt) {
            v8f acc = {};
#pragma unroll
            for (int kt = 0; kt < 4; ++kt) {
                const v16h bf = *(const v16h*)&sW2[((kt * 8 + nt) * 32 + lane) * 16];
                acc = __builtin_amdgcn_wmma_f32_16x16x32_f16(
                    false, aH[kt], false, bf, (short)0, acc, false, false);
            }
            const float bias = sB2[nt * 16 + r];
            const int   colb = nt * 16 + r;
#pragma unroll
            for (int v = 0; v < 8; ++v) {
                const int row = v + rowo;                       // C/D: lane = col, vgpr = row
                stage[row * FEAT + colb] = (_Float16)ssp(acc[v] + bias);
            }
        }

        // ---- transpose via wave-private LDS: reload filters as A fragments ----
        v16h aF[4];
#pragma unroll
        for (int kt = 0; kt < 4; ++kt) {
            const _Float16* p0 = &stage[r * FEAT + kt * 32 + hioff];
            v8h lo = *(const v8h*)p0;            // K = kt*32+hioff .. +7
            v8h hi = *(const v8h*)(p0 + 16);     // K = kt*32+16+hioff .. +7
            aF[kt] = __builtin_shufflevector(lo, hi,
                0, 1, 2, 3, 4, 5, 6, 7, 8, 9, 10, 11, 12, 13, 14, 15);
        }

        // ---- GEMM2: Y = filters @ Wd + bd; stream to HBM ----
#pragma unroll
        for (int nt = 0; nt < 8; ++nt) {
            v8f acc = {};
#pragma unroll
            for (int kt = 0; kt < 4; ++kt) {
                const v16h bf = *(const v16h*)&sWD[((kt * 8 + nt) * 32 + lane) * 16];
                acc = __builtin_amdgcn_wmma_f32_16x16x32_f16(
                    false, aF[kt], false, bf, (short)0, acc, false, false);
            }
            const float bias = sBD[nt * 16 + r];
            const int   col  = nt * 16 + r;
#pragma unroll
            for (int v = 0; v < 8; ++v) {
                const size_t row = (size_t)(t * 16 + v + rowo);
                out[row * FEAT + col] = acc[v] + bias;
            }
        }
    }
}

extern "C" void kernel_launch(void* const* d_in, const int* in_sizes, int n_in,
                              void* d_out, int out_size, void* d_ws, size_t ws_size,
                              hipStream_t stream) {
    (void)in_sizes; (void)n_in; (void)out_size; (void)d_ws; (void)ws_size;
    const float* positions = (const float*)d_in[0];
    // d_in[1] = batch_idx (int32) — unused: graphs are equal-size, ids are implicit
    const float* w1 = (const float*)d_in[2];
    const float* b1 = (const float*)d_in[3];
    const float* w2 = (const float*)d_in[4];
    const float* b2 = (const float*)d_in[5];
    const float* wd = (const float*)d_in[6];
    const float* bd = (const float*)d_in[7];
    float* out = (float*)d_out;

    dim3 grid(1024);   // 8 waves/block -> 8192 waves, 4 row-tiles per wave
    dim3 block(256);
    cfconv_fused_wmma<<<grid, block, 0, stream>>>(positions, w1, b1, w2, b2, wd, bd, out);
}
